// PointNet2FeatureExtractor_46136538693650
// MI455X (gfx1250) — compile-verified, matrix operand order checked by reading, not run on hardware
//
#include <hip/hip_runtime.h>
#include <hip/hip_bf16.h>

// ---------------------------------------------------------------------------
// Types for WMMA fragments (gfx1250, wave32)
// ---------------------------------------------------------------------------
typedef __attribute__((ext_vector_type(16))) _Float16 v16h_t;
typedef __attribute__((ext_vector_type(8)))  _Float16 v8h_t;
typedef __attribute__((ext_vector_type(8)))  float    v8f_t;

union Frag16 { v16h_t v; v8h_t h[2]; };

// 16-bit A (16x32) / B(=W row-major, i.e. B^T col-major) fragment loader.
// Per ISA layout: lane<16 holds row r, K={0..7,16..23}; lane>=16 holds row r,
// K={8..15,24..31}. Rows must be 16B aligned (stride multiple of 8 halfs).
__device__ inline v16h_t load_frag16(const _Float16* base, int stride, int lane) {
  const int r  = lane & 15;
  const int hi = (lane >> 4) & 1;
  const _Float16* p = base + (size_t)r * stride + hi * 8;
  Frag16 f;
  f.h[0] = *(const v8h_t*)(p);
  f.h[1] = *(const v8h_t*)(p + 16);
  return f.v;
}

__device__ inline v8f_t wmma_f16(v16h_t a, v16h_t b, v8f_t c) {
  return __builtin_amdgcn_wmma_f32_16x16x32_f16(false, a, false, b, (short)0, c,
                                                false, false);
}

__device__ inline unsigned ballot32(bool p) {
  return (unsigned)__ballot(p);  // wave32: low 32 bits valid
}

#define FPS_EPS 1e-5f

// ---------------------------------------------------------------------------
// Farthest point sampling: one block per batch; dist array LDS-resident.
// Matches reference scan semantics: emits far BEFORE update; starts at 0;
// argmax with first-index tie-break.
// ---------------------------------------------------------------------------
__global__ __launch_bounds__(256) void fps_kernel(const float* __restrict__ xyz,
                                                  int N, int stride, int npoint,
                                                  int* __restrict__ out) {
  __shared__ float dist[4096];
  __shared__ float rv[256];
  __shared__ int   ri[256];
  __shared__ int   s_far;
  const int b = blockIdx.x, tid = threadIdx.x;
  for (int j = tid; j < N; j += 256) dist[j] = 1e10f;
  if (tid == 0) s_far = 0;
  __syncthreads();
  for (int it = 0; it < npoint; ++it) {
    const int far = s_far;
    if (tid == 0) out[b * npoint + it] = far;
    const float* cp = xyz + ((size_t)b * N + far) * stride;
    const float cx = cp[0], cy = cp[1], cz = cp[2];
    float bv = -1.f; int bi = 0;
    for (int j = tid; j < N; j += 256) {
      const float* q = xyz + ((size_t)b * N + j) * stride;
      const float dx = q[0] - cx, dy = q[1] - cy, dz = q[2] - cz;
      const float d = dx * dx + dy * dy + dz * dz;
      float nd = dist[j];
      nd = d < nd ? d : nd;
      dist[j] = nd;
      if (nd > bv) { bv = nd; bi = j; }  // ascending j -> first max kept
    }
    rv[tid] = bv; ri[tid] = bi;
    __syncthreads();
    for (int off = 128; off; off >>= 1) {
      if (tid < off) {
        if (rv[tid + off] > rv[tid] ||
            (rv[tid + off] == rv[tid] && ri[tid + off] < ri[tid])) {
          rv[tid] = rv[tid + off]; ri[tid] = ri[tid + off];
        }
      }
      __syncthreads();
    }
    if (tid == 0) s_far = ri[0];
    __syncthreads();
  }
}

// new_xyz[b,s,:] = xyz[b, idx[b,s], 0:3]
__global__ void gather_xyz_kernel(const float* __restrict__ xyz, int stride,
                                  const int* __restrict__ idx,
                                  float* __restrict__ out, int Bn, int N, int S) {
  const int e = blockIdx.x * blockDim.x + threadIdx.x;
  if (e >= Bn * S * 3) return;
  const int c = e % 3, bs = e / 3;
  const int b = bs / S;
  out[e] = xyz[((size_t)b * N + idx[bs]) * stride + c];
}

// W (cout x cin, f32) -> Wp (coutp x cinp, f16, zero padded)
__global__ void prep_weight_kernel(const float* __restrict__ W,
                                   _Float16* __restrict__ Wp,
                                   int cout, int cin, int coutp, int cinp) {
  const int e = blockIdx.x * blockDim.x + threadIdx.x;
  if (e >= coutp * cinp) return;
  const int o = e / cinp, c = e % cinp;
  const float v = (o < cout && c < cin) ? W[(size_t)o * cin + c] : 0.f;
  Wp[e] = (_Float16)v;
}

// ---------------------------------------------------------------------------
// Fused SA-MSG scale kernel: block = one (b,s) group.
//   ball-query (ordered compaction) -> grouped f16 rows in LDS ->
//   3x (WMMA GEMM + BN + ReLU) ping-ponged in LDS -> maxpool over K -> HBM.
// ---------------------------------------------------------------------------
struct LayerP {
  const _Float16* W;           // cout x cin (f16, padded)
  const float *b, *g, *be, *m, *v;
  int cout;
};

struct SAParams {
  const float* xyz;  int xstride;            // (B,N,xstride), coords at 0..2
  const float* feats; int fstride; int Cf;   // point features
  const float* newxyz;                       // (B,S,3)
  float r2; int K; int N; int S;
  int cinp;                                  // padded input channels
  int maxch;                                 // LDS buffer channel width
  LayerP L[3];
  float* out; int Ctot; int co;              // pooled output (B,S,Ctot)
};

__global__ __launch_bounds__(256) void sa_fused_kernel(SAParams P) {
  extern __shared__ char smem[];
  const int tid  = threadIdx.x;
  const int lane = tid & 31;
  const int wave = tid >> 5;
  const int gid  = blockIdx.x;
  const int b = gid / P.S, s = gid % P.S;

  _Float16* bufA = (_Float16*)smem;
  _Float16* bufB = bufA + (size_t)P.K * P.maxch;
  int* s_idx = (int*)(bufB + (size_t)P.K * P.maxch);
  __shared__ int s_cnt;
  __shared__ int s_wc[8];

  const float cx = P.newxyz[((size_t)b * P.S + s) * 3 + 0];
  const float cy = P.newxyz[((size_t)b * P.S + s) * 3 + 1];
  const float cz = P.newxyz[((size_t)b * P.S + s) * 3 + 2];

  // ---- ball query: first K ascending indices with d2 <= r2 ----
  if (tid == 0) s_cnt = 0;
  __syncthreads();
  for (int base = 0; base < P.N; base += 256) {
    const int j = base + tid;
    bool pred = false;
    if (j < P.N) {
      const float* q = P.xyz + ((size_t)b * P.N + j) * P.xstride;
      const float dx = q[0] - cx, dy = q[1] - cy, dz = q[2] - cz;
      pred = (dx * dx + dy * dy + dz * dz) <= P.r2;
    }
    const unsigned mask = ballot32(pred);
    const int lpre = __popc(mask & ((1u << lane) - 1u));
    if (lane == 0) s_wc[wave] = __popc(mask);
    __syncthreads();
    int woff = 0;
    for (int w = 0; w < wave; ++w) woff += s_wc[w];
    const int pos = s_cnt + woff + lpre;
    if (pred && pos < P.K) s_idx[pos] = j;
    __syncthreads();
    if (tid == 0) {
      int tot = 0;
      for (int w = 0; w < 8; ++w) tot += s_wc[w];
      s_cnt += tot;
    }
    __syncthreads();
    if (s_cnt >= P.K) break;
  }
  const int cnt = s_cnt < P.K ? s_cnt : P.K;   // >=1: center is a sample
  for (int k = cnt + tid; k < P.K; k += 256) s_idx[k] = s_idx[0];
  __syncthreads();

  // ---- build grouped rows: [feats(idx) | xyz(idx)-center | 0 pad] as f16 ----
  const int cinp = P.cinp;
  for (int e = tid; e < P.K * cinp; e += 256) {
    const int k = e / cinp, c = e % cinp;
    const int j = s_idx[k];
    float val = 0.f;
    if (c < P.Cf) {
      val = P.feats[((size_t)b * P.N + j) * P.fstride + c];
    } else if (c < P.Cf + 3) {
      const int cc = c - P.Cf;
      const float ctr = (cc == 0) ? cx : (cc == 1 ? cy : cz);
      val = P.xyz[((size_t)b * P.N + j) * P.xstride + cc] - ctr;
    }
    bufA[e] = (_Float16)val;
  }
  __syncthreads();

  // ---- 3 fused WMMA layers (LDS ping-pong) ----
  _Float16* X = bufA;
  _Float16* Y = bufB;
  int cin = cinp;
  for (int l = 0; l < 3; ++l) {
    const LayerP LP = P.L[l];
    const int mt = P.K >> 4, ot = LP.cout >> 4;
    for (int t = wave; t < mt * ot; t += 8) {
      const int mi = t % mt, oi = t / mt;
      v8f_t acc = {};
      for (int k0 = 0; k0 < cin; k0 += 32) {
        const v16h_t a  = load_frag16(X + (size_t)mi * 16 * cin + k0, cin, lane);
        const v16h_t bb = load_frag16(LP.W + (size_t)oi * 16 * cin + k0, cin, lane);
        acc = wmma_f16(a, bb, acc);
      }
      const int o = oi * 16 + (lane & 15);
      const float sc = LP.g[o] * __frsqrt_rn(LP.v[o] + FPS_EPS);
      const float sh = (LP.b[o] - LP.m[o]) * sc + LP.be[o];
      const int rbase = mi * 16 + ((lane >> 4) << 3);
      #pragma unroll
      for (int i = 0; i < 8; ++i) {
        float yv = acc[i] * sc + sh;
        yv = yv > 0.f ? yv : 0.f;
        Y[(size_t)(rbase + i) * LP.cout + o] = (_Float16)yv;
      }
    }
    __syncthreads();
    _Float16* tp = X; X = Y; Y = tp;
    cin = LP.cout;
  }

  // ---- maxpool over K group members ----
  for (int c = tid; c < cin; c += 256) {
    float mx = 0.f;  // post-ReLU values are >= 0
    for (int k = 0; k < P.K; ++k) {
      const float vv = (float)X[(size_t)k * cin + c];
      mx = vv > mx ? vv : mx;
    }
    P.out[((size_t)b * P.S + s) * P.Ctot + P.co + c] = mx;
  }
}

// ---------------------------------------------------------------------------
// Small staged GEMM (sa3 layers + FC head): one wave per 16x16 tile.
// mode 1: BN+ReLU epilogue, f16 out (ld = cout). mode 0: +bias, guarded f32 out.
// ---------------------------------------------------------------------------
__global__ __launch_bounds__(32) void gemm_small_kernel(
    const _Float16* __restrict__ X, const _Float16* __restrict__ W,
    const float* __restrict__ bias, const float* __restrict__ g,
    const float* __restrict__ be, const float* __restrict__ m,
    const float* __restrict__ v, _Float16* __restrict__ Yh,
    float* __restrict__ Yf, int cinp, int cout, int store_cols, int mode) {
  const int mi = blockIdx.x, oi = blockIdx.y;
  const int lane = threadIdx.x;
  v8f_t acc = {};
  for (int k0 = 0; k0 < cinp; k0 += 32) {
    const v16h_t a  = load_frag16(X + (size_t)mi * 16 * cinp + k0, cinp, lane);
    const v16h_t bb = load_frag16(W + (size_t)oi * 16 * cinp + k0, cinp, lane);
    acc = wmma_f16(a, bb, acc);
  }
  const int o = oi * 16 + (lane & 15);
  const int rbase = mi * 16 + ((lane >> 4) << 3);
  if (mode) {
    const float sc = g[o] * __frsqrt_rn(v[o] + FPS_EPS);
    const float sh = (bias[o] - m[o]) * sc + be[o];
    #pragma unroll
    for (int i = 0; i < 8; ++i) {
      float yv = acc[i] * sc + sh;
      yv = yv > 0.f ? yv : 0.f;
      Yh[(size_t)(rbase + i) * cout + o] = (_Float16)yv;
    }
  } else {
    #pragma unroll
    for (int i = 0; i < 8; ++i) {
      if (o < store_cols) {
        const float bo = (o < store_cols) ? bias[o] : 0.f;
        Yf[(size_t)(rbase + i) * store_cols + o] = acc[i] + bo;
      }
    }
  }
}

// sa3 group-all input: rows = B*128, cols 672: [l2x(3) | l2p(640) | pad]
__global__ void build_g3_kernel(const float* __restrict__ l2x,
                                const float* __restrict__ l2p,
                                _Float16* __restrict__ Xg) {
  const int e = blockIdx.x * blockDim.x + threadIdx.x;
  const int total = 16 * 128 * 672;
  if (e >= total) return;
  const int row = e / 672, c = e % 672;
  float val = 0.f;
  if (c < 3) val = l2x[(size_t)row * 3 + c];
  else if (c < 643) val = l2p[(size_t)row * 640 + (c - 3)];
  Xg[e] = (_Float16)val;
}

// maxpool over 128 points per batch: (16*128,1024) -> (16,1024) f16
__global__ void maxpool_g3_kernel(const _Float16* __restrict__ Y,
                                  _Float16* __restrict__ feat) {
  const int e = blockIdx.x * blockDim.x + threadIdx.x;
  if (e >= 16 * 1024) return;
  const int b = e / 1024, c = e % 1024;
  float mx = 0.f;
  for (int n = 0; n < 128; ++n) {
    const float vv = (float)Y[((size_t)b * 128 + n) * 1024 + c];
    mx = vv > mx ? vv : mx;
  }
  feat[e] = (_Float16)mx;
}

// ---------------------------------------------------------------------------
// Host launcher
// ---------------------------------------------------------------------------
struct ConvRaw { const float *W, *b, *g, *be, *m, *v; };

extern "C" void kernel_launch(void* const* d_in, const int* in_sizes, int n_in,
                              void* d_out, int out_size, void* d_ws, size_t ws_size,
                              hipStream_t stream) {
  (void)in_sizes; (void)n_in; (void)out_size; (void)ws_size;

  // ---- walk inputs in setup_inputs() insertion order ----
  int ii = 0;
  const float* pc = (const float*)d_in[ii++];  // (16,4096,6)
  ConvRaw sa1[3][3], sa2[3][3], sa3[3];
  for (int s = 0; s < 3; ++s)
    for (int l = 0; l < 3; ++l) {
      ConvRaw& c = sa1[s][l];
      c.W = (const float*)d_in[ii++]; c.b = (const float*)d_in[ii++];
      c.g = (const float*)d_in[ii++]; c.be = (const float*)d_in[ii++];
      c.m = (const float*)d_in[ii++]; c.v = (const float*)d_in[ii++];
    }
  for (int s = 0; s < 3; ++s)
    for (int l = 0; l < 3; ++l) {
      ConvRaw& c = sa2[s][l];
      c.W = (const float*)d_in[ii++]; c.b = (const float*)d_in[ii++];
      c.g = (const float*)d_in[ii++]; c.be = (const float*)d_in[ii++];
      c.m = (const float*)d_in[ii++]; c.v = (const float*)d_in[ii++];
    }
  for (int l = 0; l < 3; ++l) {
    ConvRaw& c = sa3[l];
    c.W = (const float*)d_in[ii++]; c.b = (const float*)d_in[ii++];
    c.g = (const float*)d_in[ii++]; c.be = (const float*)d_in[ii++];
    c.m = (const float*)d_in[ii++]; c.v = (const float*)d_in[ii++];
  }
  const float* fc1W = (const float*)d_in[ii++];
  const float* fc1b = (const float*)d_in[ii++];
  const float* bn1g = (const float*)d_in[ii++];
  const float* bn1be = (const float*)d_in[ii++];
  const float* bn1m = (const float*)d_in[ii++];
  const float* bn1v = (const float*)d_in[ii++];
  const float* fc2W = (const float*)d_in[ii++];
  const float* fc2b = (const float*)d_in[ii++];

  // ---- workspace bump allocator ----
  char* wsb = (char*)d_ws;
  size_t off = 0;
  auto walloc = [&](size_t bytes) -> void* {
    void* p = wsb + off;
    off = (off + bytes + 255) & ~(size_t)255;
    return p;
  };

  const int Bn = 16;
  int* fps1 = (int*)walloc(Bn * 512 * sizeof(int));
  float* l1x = (float*)walloc((size_t)Bn * 512 * 3 * sizeof(float));
  float* l1p = (float*)walloc((size_t)Bn * 512 * 320 * sizeof(float));
  int* fps2 = (int*)walloc(Bn * 128 * sizeof(int));
  float* l2x = (float*)walloc((size_t)Bn * 128 * 3 * sizeof(float));
  float* l2p = (float*)walloc((size_t)Bn * 128 * 640 * sizeof(float));
  _Float16* Xg = (_Float16*)walloc((size_t)Bn * 128 * 672 * sizeof(_Float16));
  _Float16* Y1 = (_Float16*)walloc((size_t)Bn * 128 * 256 * sizeof(_Float16));
  _Float16* Y2 = (_Float16*)walloc((size_t)Bn * 128 * 512 * sizeof(_Float16));
  _Float16* Y3 = (_Float16*)walloc((size_t)Bn * 128 * 1024 * sizeof(_Float16));
  _Float16* feat_h = (_Float16*)walloc((size_t)Bn * 1024 * sizeof(_Float16));
  _Float16* fc1h = (_Float16*)walloc((size_t)Bn * 512 * sizeof(_Float16));

  const int sa1_w[3][3] = {{32, 32, 64}, {64, 64, 128}, {64, 96, 128}};
  const int sa2_w[3][3] = {{64, 64, 128}, {128, 128, 256}, {128, 128, 256}};
  const int sa3_w[3] = {256, 512, 1024};
  const int   sa1_K[3] = {16, 32, 128};
  const float sa1_r2[3] = {0.01f, 0.04f, 0.16f};
  const int   sa2_K[3] = {32, 64, 128};
  const float sa2_r2[3] = {0.04f, 0.16f, 0.64f};
  const int sa1_co[3] = {0, 64, 192};
  const int sa2_co[3] = {0, 128, 384};

  auto prepW = [&](const float* W, int cout, int cin, int cinp) -> _Float16* {
    _Float16* Wp = (_Float16*)walloc((size_t)cout * cinp * sizeof(_Float16));
    const int n = cout * cinp;
    prep_weight_kernel<<<(n + 255) / 256, 256, 0, stream>>>(W, Wp, cout, cin,
                                                            cout, cinp);
    return Wp;
  };

  // prep all conv weights (first layer of each MLP pads cin up to x32)
  _Float16 *sa1Wp[3][3], *sa2Wp[3][3], *sa3Wp[3];
  for (int s = 0; s < 3; ++s) {
    int cin = 6, cinp = 32;
    for (int l = 0; l < 3; ++l) {
      sa1Wp[s][l] = prepW(sa1[s][l].W, sa1_w[s][l], cin, cinp);
      cin = sa1_w[s][l]; cinp = cin;
    }
  }
  for (int s = 0; s < 3; ++s) {
    int cin = 323, cinp = 352;
    for (int l = 0; l < 3; ++l) {
      sa2Wp[s][l] = prepW(sa2[s][l].W, sa2_w[s][l], cin, cinp);
      cin = sa2_w[s][l]; cinp = cin;
    }
  }
  {
    int cin = 643, cinp = 672;
    for (int l = 0; l < 3; ++l) {
      sa3Wp[l] = prepW(sa3[l].W, sa3_w[l], cin, cinp);
      cin = sa3_w[l]; cinp = cin;
    }
  }
  _Float16* fc1Wp = prepW(fc1W, 512, 1024, 1024);
  _Float16* fc2Wp = (_Float16*)walloc((size_t)256 * 512 * sizeof(_Float16));
  {
    const int n = 256 * 512;
    prep_weight_kernel<<<(n + 255) / 256, 256, 0, stream>>>(fc2W, fc2Wp, 248,
                                                            512, 256, 512);
  }

  // ---- SA1 ----
  fps_kernel<<<Bn, 256, 0, stream>>>(pc, 4096, 6, 512, fps1);
  {
    const int n = Bn * 512 * 3;
    gather_xyz_kernel<<<(n + 255) / 256, 256, 0, stream>>>(pc, 6, fps1, l1x, Bn,
                                                           4096, 512);
  }
  for (int s = 0; s < 3; ++s) {
    SAParams P;
    P.xyz = pc; P.xstride = 6;
    P.feats = pc + 3; P.fstride = 6; P.Cf = 3;
    P.newxyz = l1x;
    P.r2 = sa1_r2[s]; P.K = sa1_K[s]; P.N = 4096; P.S = 512;
    P.cinp = 32;
    int maxch = 32;
    int cin = 32;
    for (int l = 0; l < 3; ++l) {
      P.L[l].W = sa1Wp[s][l]; P.L[l].b = sa1[s][l].b; P.L[l].g = sa1[s][l].g;
      P.L[l].be = sa1[s][l].be; P.L[l].m = sa1[s][l].m; P.L[l].v = sa1[s][l].v;
      P.L[l].cout = sa1_w[s][l];
      if (sa1_w[s][l] > maxch) maxch = sa1_w[s][l];
      cin = sa1_w[s][l];
    }
    (void)cin;
    P.maxch = maxch;
    P.out = l1p; P.Ctot = 320; P.co = sa1_co[s];
    const size_t smem = (size_t)2 * P.K * maxch * sizeof(_Float16) +
                        (size_t)P.K * sizeof(int) + 64;
    hipFuncSetAttribute((const void*)sa_fused_kernel,
                        hipFuncAttributeMaxDynamicSharedMemorySize, (int)smem);
    sa_fused_kernel<<<Bn * 512, 256, smem, stream>>>(P);
  }

  // ---- SA2 ----
  fps_kernel<<<Bn, 256, 0, stream>>>(l1x, 512, 3, 128, fps2);
  {
    const int n = Bn * 128 * 3;
    gather_xyz_kernel<<<(n + 255) / 256, 256, 0, stream>>>(l1x, 3, fps2, l2x, Bn,
                                                           512, 128);
  }
  for (int s = 0; s < 3; ++s) {
    SAParams P;
    P.xyz = l1x; P.xstride = 3;
    P.feats = l1p; P.fstride = 320; P.Cf = 320;
    P.newxyz = l2x;
    P.r2 = sa2_r2[s]; P.K = sa2_K[s]; P.N = 512; P.S = 128;
    P.cinp = 352;
    int maxch = 352;
    for (int l = 0; l < 3; ++l) {
      P.L[l].W = sa2Wp[s][l]; P.L[l].b = sa2[s][l].b; P.L[l].g = sa2[s][l].g;
      P.L[l].be = sa2[s][l].be; P.L[l].m = sa2[s][l].m; P.L[l].v = sa2[s][l].v;
      P.L[l].cout = sa2_w[s][l];
      if (sa2_w[s][l] > maxch) maxch = sa2_w[s][l];
    }
    P.maxch = maxch;
    P.out = l2p; P.Ctot = 640; P.co = sa2_co[s];
    const size_t smem = (size_t)2 * P.K * maxch * sizeof(_Float16) +
                        (size_t)P.K * sizeof(int) + 64;
    hipFuncSetAttribute((const void*)sa_fused_kernel,
                        hipFuncAttributeMaxDynamicSharedMemorySize, (int)smem);
    sa_fused_kernel<<<Bn * 128, 256, smem, stream>>>(P);
  }

  // ---- SA3 (group all) ----
  {
    const int n = Bn * 128 * 672;
    build_g3_kernel<<<(n + 255) / 256, 256, 0, stream>>>(l2x, l2p, Xg);
  }
  const int M3 = Bn * 128;  // 2048 rows
  gemm_small_kernel<<<dim3(M3 / 16, 256 / 16), 32, 0, stream>>>(
      Xg, sa3Wp[0], sa3[0].b, sa3[0].g, sa3[0].be, sa3[0].m, sa3[0].v, Y1,
      nullptr, 672, 256, 256, 1);
  gemm_small_kernel<<<dim3(M3 / 16, 512 / 16), 32, 0, stream>>>(
      Y1, sa3Wp[1], sa3[1].b, sa3[1].g, sa3[1].be, sa3[1].m, sa3[1].v, Y2,
      nullptr, 256, 512, 512, 1);
  gemm_small_kernel<<<dim3(M3 / 16, 1024 / 16), 32, 0, stream>>>(
      Y2, sa3Wp[2], sa3[2].b, sa3[2].g, sa3[2].be, sa3[2].m, sa3[2].v, Y3,
      nullptr, 512, 1024, 1024, 1);
  {
    const int n = Bn * 1024;
    maxpool_g3_kernel<<<(n + 255) / 256, 256, 0, stream>>>(Y3, feat_h);
  }

  // ---- FC head ----
  gemm_small_kernel<<<dim3(Bn / 16, 512 / 16), 32, 0, stream>>>(
      feat_h, fc1Wp, fc1b, bn1g, bn1be, bn1m, bn1v, fc1h, nullptr, 1024, 512,
      512, 1);
  gemm_small_kernel<<<dim3(Bn / 16, 256 / 16), 32, 0, stream>>>(
      fc1h, fc2Wp, fc2b, nullptr, nullptr, nullptr, nullptr, nullptr,
      (float*)d_out, 512, 256, 248, 0);
}
